// BaseModel_75522704933527
// MI455X (gfx1250) — compile-verified, matrix operand order checked by reading, not run on hardware
//
#include <hip/hip_runtime.h>
#include <hip/hip_bf16.h>

#define N_ATOMS  5000
#define N_PAIRS  160000
#define K_CH     128
#define FEAT_F   1152   // 9 * 128 floats per atom (f0:128, f1:384, f2:640)

typedef _Float16     v16h  __attribute__((ext_vector_type(16)));
typedef float        v8f   __attribute__((ext_vector_type(8)));
typedef unsigned int u32x4 __attribute__((ext_vector_type(4)));
typedef int          i32x4 __attribute__((ext_vector_type(4)));
typedef int          i32x8 __attribute__((ext_vector_type(8)));

#if defined(__has_builtin)
#  if __has_builtin(__builtin_amdgcn_tensor_load_to_lds)
#    define HAVE_TDM 1
#  endif
#endif
#ifndef HAVE_TDM
#  define HAVE_TDM 0
#endif

__device__ __forceinline__ float silu_f(float x) { return x / (1.0f + __expf(-x)); }

__device__ __forceinline__ v8f wmma_f16(v16h a, v16h b, v8f c) {
  // D = A(16x32 f16) * B(32x16 f16) + C(16x16 f32)
  return __builtin_amdgcn_wmma_f32_16x16x32_f16(false, a, false, b, (short)0, c, false, false);
}

// WMMA 16-bit fragment swizzle (ISA 7.12.2): element (k, col) of a K x N
// matrix lives in k-window ks=k>>5, column tile nt=col>>4, lane
// (col&15) + 16*((k>>3)&1), half (k&7) + 8*((k>>4)&1).
__device__ __forceinline__ int frag_idx(int k, int col, int ntiles) {
  int ks = k >> 5, kloc = k & 31;
  int nt = col >> 4, n = col & 15;
  int lane = n + 16 * ((kloc >> 3) & 1);
  int i = (kloc & 7) + 8 * ((kloc >> 4) & 1);
  return ((ks * ntiles + nt) * 32 + lane) * 16 + i;
}

#if HAVE_TDM
// Issue a TDM DMA of a contiguous f16 block (nelem elements) into LDS.
// D# built per CDNA5 ISA ch.8 (1-D tile: tile_dim0 = nelem <= 65535).
__device__ __forceinline__ void tdm_load_f16(void* lds_dst, const void* gsrc, unsigned nelem) {
  unsigned long long ga = (unsigned long long)(uintptr_t)gsrc;
  unsigned la = (unsigned)(uintptr_t)lds_dst;  // flat LDS addr, low 32 = byte offset
  u32x4 g0;
  g0[0] = 1u;                                 // count=1, is_restore=0, no gather
  g0[1] = la;                                 // lds_addr
  g0[2] = (unsigned)ga;                       // global_addr[31:0]
  g0[3] = (unsigned)(ga >> 32) | (2u << 30);  // global_addr[56:32] | type=2
  i32x8 g1;
  g1[0] = (int)(1u << 16);                    // workgroup_mask=0, data_size=1 (2B)
  g1[1] = (int)(nelem << 16);                 // tensor_dim0[15:0] at [63:48]
  g1[2] = (int)((nelem >> 16) | (1u << 16));  // tensor_dim0 hi | tensor_dim1=1
  g1[3] = (int)(nelem << 16);                 // tensor_dim1 hi=0 | tile_dim0=nelem
  g1[4] = 1;                                  // tile_dim1=1 | tile_dim2=0
  g1[5] = (int)nelem;                         // tensor_dim0_stride[31:0]
  g1[6] = 0;
  g1[7] = 0;
  i32x4 z4 = {0, 0, 0, 0};
#if __clang_major__ >= 23
  i32x8 z8 = {0, 0, 0, 0, 0, 0, 0, 0};
  __builtin_amdgcn_tensor_load_to_lds(g0, g1, z4, z4, z8, 0);
#else
  __builtin_amdgcn_tensor_load_to_lds(g0, g1, z4, z4, 0);
#endif
}
#endif

// Stage a pre-swizzled weight block into LDS: TDM DMA (wave 0) if available,
// otherwise cooperative copy. Caller barriers before use.
__device__ __forceinline__ void stage_issue(_Float16* lds_dst, const _Float16* gsrc, int nelem) {
#if HAVE_TDM
  if (threadIdx.x < 32) tdm_load_f16(lds_dst, gsrc, (unsigned)nelem);
#else
  for (int e = threadIdx.x; e < nelem; e += blockDim.x) lds_dst[e] = gsrc[e];
#endif
}

__device__ __forceinline__ void stage_wait() {
#if HAVE_TDM
  if (threadIdx.x < 32) __builtin_amdgcn_s_wait_tensorcnt(0);
#endif
}

// ---------------------------------------------------------------- utilities
__global__ __launch_bounds__(256) void zero_kernel(float* p, int n) {
  int i = blockIdx.x * 256 + threadIdx.x;
  if (i < n) p[i] = 0.0f;
}

// Convert K x N f32 (row-major) weights to f16 packed in WMMA B-fragment order.
__global__ __launch_bounds__(256)
void pack_b16_kernel(const float* __restrict__ src, _Float16* __restrict__ dst, int K, int N) {
  int e = blockIdx.x * 256 + threadIdx.x;
  if (e >= K * N) return;
  int k = e / N, col = e - k * N;
  dst[frag_idx(k, col, N >> 4)] = (_Float16)src[e];
}

// ---------------------------------------------------------------- pair geometry
__global__ __launch_bounds__(256)
void geom_kernel(const float* __restrict__ pos, const float* __restrict__ cells,
                 const int* __restrict__ shifts, const int* __restrict__ ctr,
                 const int* __restrict__ nbr, const int* __restrict__ spair,
                 float* __restrict__ geom) {
  int p = blockIdx.x * 256 + threadIdx.x;
  if (p >= N_PAIRS) return;
  int ci = ctr[p], ni = nbr[p], si = spair[p];
  float s0 = (float)shifts[p * 3 + 0] - 1.0f;
  float s1 = (float)shifts[p * 3 + 1] - 1.0f;
  float s2 = (float)shifts[p * 3 + 2] - 1.0f;
  const float* C = cells + si * 9;
  float v[3];
  #pragma unroll
  for (int j = 0; j < 3; ++j)
    v[j] = pos[ni * 3 + j] - pos[ci * 3 + j] + s0 * C[0 * 3 + j] + s1 * C[1 * 3 + j] + s2 * C[2 * 3 + j];
  float d = sqrtf(v[0] * v[0] + v[1] * v[1] + v[2] * v[2] + 1e-12f);
  float inv = 1.0f / d;
  geom[p * 4 + 0] = d;
  geom[p * 4 + 1] = v[0] * inv;
  geom[p * 4 + 2] = v[1] * inv;
  geom[p * 4 + 3] = v[2] * inv;
}

// ---------------------------------------------------------------- message pass
// One block = 16 pairs. TDM stages the pre-swizzled 32x384 f16 weights into LDS
// while VALU computes radial layer 1 (written directly in A-fragment order);
// 8 waves run 24 WMMA tiles with single-vector LDS fragment loads; then
// 16x9x128 atomic adds scatter messages into L2-resident per-atom features.
__global__ __launch_bounds__(256)
void msg_kernel(const float* __restrict__ geom,
                const int* __restrict__ ctr, const int* __restrict__ nbr,
                const int* __restrict__ species, const float* __restrict__ embed,
                const float* __restrict__ w1, const float* __restrict__ b1,
                const _Float16* __restrict__ w2p, const float* __restrict__ b2,
                float* __restrict__ f0o, float* __restrict__ f1o, float* __restrict__ f2o,
                const float* __restrict__ f0i, const float* __restrict__ f1i,
                const float* __restrict__ f2i, int pass) {
  __shared__ v16h  WldsV[24 * 32];     // staged swizzled weights (24 KB)
  __shared__ v16h  AhV[32];            // layer-1 activations in A-fragment order (1 KB)
  __shared__ float RkL[16][384];       // radial outputs, all 3 l-blocks
  __shared__ float shL[16][9];         // spherical harmonics * INIT_SCALE
  __shared__ float dL[16];
  __shared__ int   ctrL[16], nbrL[16], embRowL[16];

  const int tid = threadIdx.x;
  const int p0 = blockIdx.x * 16;
  _Float16* AhH = (_Float16*)AhV;

  stage_issue((_Float16*)WldsV, w2p, 32 * 384);  // async DMA, overlapped with phases below

  if (tid < 16) {
    int p = p0 + tid;
    float d = geom[p * 4 + 0], ux = geom[p * 4 + 1], uy = geom[p * 4 + 2], uz = geom[p * 4 + 3];
    dL[tid] = d;
    const float IS = 0.2f;                       // INIT_SCALE
    const float c3 = 1.7320508075688772f;        // sqrt(3)
    shL[tid][0] = IS;
    shL[tid][1] = uy * IS; shL[tid][2] = uz * IS; shL[tid][3] = ux * IS;
    shL[tid][4] = c3 * ux * uy * IS;
    shL[tid][5] = c3 * uy * uz * IS;
    shL[tid][6] = 0.5f * (3.0f * uz * uz - 1.0f) * IS;
    shL[tid][7] = c3 * ux * uz * IS;
    shL[tid][8] = 0.5f * c3 * (ux * ux - uy * uy) * IS;
    int nb = nbr[p];
    ctrL[tid] = ctr[p]; nbrL[tid] = nb; embRowL[tid] = species[nb];
  }
  __syncthreads();

  // Radial layer 1: 16 pairs x 32 hidden; store in WMMA A-fragment order.
  for (int e = tid; e < 512; e += 256) {
    int p = e >> 5, j = e & 31;
    float d = dL[p];
    float fcut = (d < 5.0f) ? 0.5f * (__cosf(3.14159265358979323f * d * 0.2f) + 1.0f) : 0.0f;
    float acc = b1[j];
    #pragma unroll
    for (int bb = 0; bb < 8; ++bb) {
      float cb = (5.0f / 7.0f) * (float)bb;      // linspace(0, 5, 8)
      float dd = d - cb;
      float x = __expf(-2.0f * dd * dd) * fcut;  // 1/(2*0.5^2) = 2
      acc += x * w1[bb * 32 + j];
    }
    int lane = p + 16 * ((j >> 3) & 1);
    int half = (j & 7) + 8 * ((j >> 4) & 1);
    AhH[lane * 16 + half] = (_Float16)silu_f(acc);
  }
  __syncthreads();
  stage_wait();          // wave 0 drains TENSORcnt
  __syncthreads();       // weights visible to all waves

  // Radial layer 2 via WMMA: 24 tiles, fragments loaded as single 32B vectors.
  {
    int wave = tid >> 5, lane = tid & 31;
    int n = lane & 15, mb = (lane >> 4) * 8;
    v16h a = AhV[lane];
    #pragma unroll
    for (int t = 0; t < 3; ++t) {
      int nt = wave * 3 + t, col0 = nt * 16;
      v16h bv = WldsV[nt * 32 + lane];
      v8f c = {};
      c = wmma_f16(a, bv, c);
      float bias = b2[col0 + n];
      #pragma unroll
      for (int r = 0; r < 8; ++r) RkL[mb + r][col0 + n] = c[r] + bias;
    }
  }
  __syncthreads();

  // Scatter: 16 pairs x 9 components x 128 channels -> atomic adds into feats
  const float MS = 0.1767767f;  // MSG_SCALE
  for (int e = tid; e < 16 * 9 * 128; e += 256) {
    int kk = e & 127;
    int r2 = e >> 7;
    int p = r2 / 9;
    int c9 = r2 - p * 9;
    int l = (c9 == 0) ? 0 : ((c9 < 4) ? 1 : 2);
    int comp = (c9 == 0) ? 0 : ((c9 < 4) ? c9 - 1 : c9 - 4);
    int stride = (l == 0) ? 128 : ((l == 1) ? 384 : 640);
    float rk = RkL[p][l * 128 + kk];
    float shv = shL[p][c9];
    float val;
    if (pass == 0) {
      float emb = embed[embRowL[p] * 128 + kk];
      val = shv * rk * emb;
    } else {
      int nb = nbrL[p];
      float invn = f0i[nb * 128 + kk];
      const float* fin = (l == 0) ? f0i : ((l == 1) ? f1i : f2i);
      float g = fin[nb * stride + comp * 128 + kk];
      float rk0 = RkL[p][kk];
      val = shv * rk * invn + rk0 * g;
    }
    float* fo = (l == 0) ? f0o : ((l == 1) ? f1o : f2o);
    atomicAdd(fo + ctrL[p] * stride + comp * 128 + kk, MS * val);
  }
}

// ---------------------------------------------------------------- CG tensor-square (per atom, per channel)
__global__ __launch_bounds__(256)
void cg_kernel(float* __restrict__ f0, float* __restrict__ f1, float* __restrict__ f2,
               const float* __restrict__ mix, const float* __restrict__ U2g) {
  __shared__ float U2[81];
  int tid = threadIdx.x;
  if (tid < 81) U2[tid] = U2g[tid];
  __syncthreads();
  int g = blockIdx.x * 256 + tid;
  if (g >= N_ATOMS * K_CH) return;
  int atom = g >> 7, k = g & 127;

  { // l = 0: out = f + f^2 * mix0
    float v = f0[atom * 128 + k];
    f0[atom * 128 + k] = v + v * v * mix[k];
  }
  { // l = 1: uncouple (U2^T), A@A, couple (U2)
    float v[3];
    #pragma unroll
    for (int j = 0; j < 3; ++j) v[j] = f1[atom * 384 + j * 128 + k];
    float A9[9], M9[9];
    #pragma unroll
    for (int p = 0; p < 9; ++p) {
      float s = 0.0f;
      #pragma unroll
      for (int j = 0; j < 3; ++j) s += U2[(1 + j) * 9 + p] * v[j];
      A9[p] = s;
    }
    float mx = mix[128 + k];
    #pragma unroll
    for (int i = 0; i < 3; ++i)
      #pragma unroll
      for (int j = 0; j < 3; ++j) {
        float s = 0.0f;
        #pragma unroll
        for (int m2 = 0; m2 < 3; ++m2) s += A9[i * 3 + m2] * A9[m2 * 3 + j];
        M9[i * 3 + j] = A9[i * 3 + j] + s * mx;
      }
    #pragma unroll
    for (int j = 0; j < 3; ++j) {
      float s = 0.0f;
      #pragma unroll
      for (int q = 0; q < 9; ++q) s += U2[(1 + j) * 9 + q] * M9[q];
      f1[atom * 384 + j * 128 + k] = s;
    }
  }
  { // l = 2
    float v[5];
    #pragma unroll
    for (int j = 0; j < 5; ++j) v[j] = f2[atom * 640 + j * 128 + k];
    float A9[9], M9[9];
    #pragma unroll
    for (int p = 0; p < 9; ++p) {
      float s = 0.0f;
      #pragma unroll
      for (int j = 0; j < 5; ++j) s += U2[(4 + j) * 9 + p] * v[j];
      A9[p] = s;
    }
    float mx = mix[256 + k];
    #pragma unroll
    for (int i = 0; i < 3; ++i)
      #pragma unroll
      for (int j = 0; j < 3; ++j) {
        float s = 0.0f;
        #pragma unroll
        for (int m2 = 0; m2 < 3; ++m2) s += A9[i * 3 + m2] * A9[m2 * 3 + j];
        M9[i * 3 + j] = A9[i * 3 + j] + s * mx;
      }
    #pragma unroll
    for (int j = 0; j < 5; ++j) {
      float s = 0.0f;
      #pragma unroll
      for (int q = 0; q < 9; ++q) s += U2[(4 + j) * 9 + q] * M9[q];
      f2[atom * 640 + j * 128 + k] = s;
    }
  }
}

// ---------------------------------------------------------------- head MLP
// Two 128x128 WMMA layers + final dot. Pre-swizzled weights TDM-staged into one
// reusable 32 KB LDS buffer; activations kept in A-fragment order throughout.
__global__ __launch_bounds__(256)
void head_kernel(const float* __restrict__ f0,
                 const _Float16* __restrict__ w1p, const float* __restrict__ b1,
                 const _Float16* __restrict__ w2p, const float* __restrict__ b2,
                 const float* __restrict__ lw, const float* __restrict__ lb,
                 float* __restrict__ out) {
  __shared__ v16h  WlV[32 * 32];       // staged swizzled weights (32 KB), reused
  __shared__ v16h  A0V[4 * 32];        // 16x128 activations, A-fragment order (4 KB)
  __shared__ v16h  B1V[4 * 32];        // inter-layer activations (4 KB)
  __shared__ float H2[16][128];
  int tid = threadIdx.x;
  int tile = blockIdx.x;
  _Float16* A0H = (_Float16*)A0V;
  _Float16* B1H = (_Float16*)B1V;

  stage_issue((_Float16*)WlV, w1p, 128 * 128);   // DMA layer-1 weights

  for (int e = tid; e < 2048; e += 256) {
    int p = e >> 7, kk = e & 127;
    int row = tile * 16 + p;
    float v = (row < N_ATOMS) ? f0[row * 128 + kk] : 0.0f;
    int lane = p + 16 * (((kk & 31) >> 3) & 1);
    int half = (kk & 7) + 8 * ((kk >> 4) & 1);
    A0H[((kk >> 5) * 32 + lane) * 16 + half] = (_Float16)v;
  }
  __syncthreads();
  stage_wait();
  __syncthreads();

  int wave = tid >> 5, lane = tid & 31;
  int n = lane & 15, mb = (lane >> 4) * 8;
  int col0 = wave * 16;

  { // layer 1: 4 accumulating K-steps of 32
    v8f c = {};
    #pragma unroll
    for (int ks = 0; ks < 4; ++ks) {
      v16h a = A0V[ks * 32 + lane];
      v16h bv = WlV[(ks * 8 + wave) * 32 + lane];
      c = wmma_f16(a, bv, c);
    }
    float bias = b1[col0 + n];
    #pragma unroll
    for (int r = 0; r < 8; ++r) {
      int m = mb + r, kk = col0 + n;
      float s = silu_f(c[r] + bias);
      int lane2 = m + 16 * (((kk & 31) >> 3) & 1);
      int half = (kk & 7) + 8 * ((kk >> 4) & 1);
      B1H[((kk >> 5) * 32 + lane2) * 16 + half] = (_Float16)s;
    }
  }
  __syncthreads();          // layer-1 reads of WlV complete
  stage_issue((_Float16*)WlV, w2p, 128 * 128);   // DMA layer-2 weights, same buffer
  stage_wait();
  __syncthreads();

  { // layer 2
    v8f c = {};
    #pragma unroll
    for (int ks = 0; ks < 4; ++ks) {
      v16h a = B1V[ks * 32 + lane];
      v16h bv = WlV[(ks * 8 + wave) * 32 + lane];
      c = wmma_f16(a, bv, c);
    }
    float bias = b2[col0 + n];
    #pragma unroll
    for (int r = 0; r < 8; ++r) H2[mb + r][col0 + n] = silu_f(c[r] + bias);
  }
  __syncthreads();

  if (tid < 16) {
    int row = tile * 16 + tid;
    if (row < N_ATOMS) {
      float s = lb[0];
      for (int kk = 0; kk < 128; ++kk) s += H2[tid][kk] * lw[kk];
      out[row] = s;
    }
  }
}

// ---------------------------------------------------------------- launcher
extern "C" void kernel_launch(void* const* d_in, const int* in_sizes, int n_in,
                              void* d_out, int out_size, void* d_ws, size_t ws_size,
                              hipStream_t stream) {
  const float* positions = (const float*)d_in[0];
  const float* cells     = (const float*)d_in[1];
  const int*   species   = (const int*)d_in[2];
  const int*   shifts    = (const int*)d_in[3];
  const int*   ctr       = (const int*)d_in[4];
  const int*   nbr       = (const int*)d_in[5];
  const int*   spair     = (const int*)d_in[6];
  const float* embed     = (const float*)d_in[7];
  const float* rad_w1    = (const float*)d_in[8];
  const float* rad_b1    = (const float*)d_in[9];
  const float* rad_w2    = (const float*)d_in[10];
  const float* rad_b2    = (const float*)d_in[11];
  const float* erad_w1   = (const float*)d_in[12];
  const float* erad_b1   = (const float*)d_in[13];
  const float* erad_w2   = (const float*)d_in[14];
  const float* erad_b2   = (const float*)d_in[15];
  const float* mix_a     = (const float*)d_in[16];
  const float* emix_a    = (const float*)d_in[17];
  const float* head_w1   = (const float*)d_in[18];
  const float* head_b1   = (const float*)d_in[19];
  const float* head_w2   = (const float*)d_in[20];
  const float* head_b2   = (const float*)d_in[21];
  const float* last_w    = (const float*)d_in[22];
  const float* last_b    = (const float*)d_in[23];
  const float* U2        = (const float*)d_in[24];
  (void)in_sizes; (void)n_in; (void)out_size; (void)ws_size;

  char* ws = (char*)d_ws;
  size_t off = 0;
  auto alloc = [&](size_t bytes) -> void* {
    void* p = ws + off;
    off += (bytes + 255) & ~(size_t)255;
    return p;
  };
  float*    geom    = (float*)alloc((size_t)N_PAIRS * 4 * sizeof(float));
  float*    featsA  = (float*)alloc((size_t)N_ATOMS * FEAT_F * sizeof(float));
  float*    featsB  = (float*)alloc((size_t)N_ATOMS * FEAT_F * sizeof(float));
  _Float16* radw2p  = (_Float16*)alloc(32 * 384 * sizeof(_Float16));
  _Float16* eradw2p = (_Float16*)alloc(32 * 384 * sizeof(_Float16));
  _Float16* hw1p    = (_Float16*)alloc(128 * 128 * sizeof(_Float16));
  _Float16* hw2p    = (_Float16*)alloc(128 * 128 * sizeof(_Float16));

  float* a0 = featsA;
  float* a1 = featsA + (size_t)N_ATOMS * 128;
  float* a2 = featsA + (size_t)N_ATOMS * (128 + 384);
  float* b0 = featsB;
  float* b1p = featsB + (size_t)N_ATOMS * 128;
  float* b2p = featsB + (size_t)N_ATOMS * (128 + 384);

  // Pack weights to f16 in WMMA fragment order (once per launch; deterministic)
  pack_b16_kernel<<<(32 * 384 + 255) / 256, 256, 0, stream>>>(rad_w2, radw2p, 32, 384);
  pack_b16_kernel<<<(32 * 384 + 255) / 256, 256, 0, stream>>>(erad_w2, eradw2p, 32, 384);
  pack_b16_kernel<<<(128 * 128 + 255) / 256, 256, 0, stream>>>(head_w1, hw1p, 128, 128);
  pack_b16_kernel<<<(128 * 128 + 255) / 256, 256, 0, stream>>>(head_w2, hw2p, 128, 128);

  // Zero feature accumulators
  int nfeat = N_ATOMS * FEAT_F;
  zero_kernel<<<(nfeat + 255) / 256, 256, 0, stream>>>(featsA, nfeat);

  // Pair geometry
  geom_kernel<<<(N_PAIRS + 255) / 256, 256, 0, stream>>>(positions, cells, shifts, ctr, nbr,
                                                         spair, geom);

  // Message pass 1 (TDM-staged weights, radial MLP via WMMA, atomic scatter)
  msg_kernel<<<N_PAIRS / 16, 256, 0, stream>>>(geom, ctr, nbr, species, embed,
                                               rad_w1, rad_b1, radw2p, rad_b2,
                                               a0, a1, a2, nullptr, nullptr, nullptr, 0);

  // CG iterate 1
  cg_kernel<<<(N_ATOMS * K_CH + 255) / 256, 256, 0, stream>>>(a0, a1, a2, mix_a, U2);

  // new feats start as copy of feats
  hipMemcpyAsync(featsB, featsA, (size_t)nfeat * sizeof(float), hipMemcpyDeviceToDevice, stream);

  // Message pass 2
  msg_kernel<<<N_PAIRS / 16, 256, 0, stream>>>(geom, ctr, nbr, species, embed,
                                               erad_w1, erad_b1, eradw2p, erad_b2,
                                               b0, b1p, b2p, a0, a1, a2, 1);

  // CG iterate 2
  cg_kernel<<<(N_ATOMS * K_CH + 255) / 256, 256, 0, stream>>>(b0, b1p, b2p, emix_a, U2);

  // Head MLP (WMMA, TDM-staged weights) -> per-atom scalar
  head_kernel<<<(N_ATOMS + 15) / 16, 256, 0, stream>>>(b0, hw1p, head_b1, hw2p, head_b2,
                                                       last_w, last_b, (float*)d_out);
}
